// TransformerEncoderLayer_32366873543454
// MI455X (gfx1250) — matrix operand order repaired and checked by reading, not compiled, not run on hardware
//
#include <hip/hip_runtime.h>

// ---------------------------------------------------------------------------
// Types for CDNA5 WMMA (gfx1250, wave32)
// ---------------------------------------------------------------------------
typedef __bf16 bf16_t;
typedef bf16_t v16bf __attribute__((ext_vector_type(16)));
typedef bf16_t v8bf  __attribute__((ext_vector_type(8)));
typedef float  v8f   __attribute__((ext_vector_type(8)));
typedef int    v4i   __attribute__((ext_vector_type(4)));

static __device__ __forceinline__ bf16_t f2bf(float f) { return (bf16_t)f; }

// Problem constants
#define BB   8
#define NN   1024
#define DD   768
#define HH   12
#define DH   64
#define DFF  3072
#define MTOT (BB * NN)          // 8192 rows
#define QKVN (3 * DD)           // 2304

// ---------------------------------------------------------------------------
// Async global->LDS staging (GLOBAL_LOAD_ASYNC_TO_LDS_B128, ASYNCcnt).
// Builtin signature (from hipcc diagnostic): param0 = AS(1) int4*,
// param1 = AS(3) int4*, then imm offset, imm cpol.
// Guarded by __has_builtin; fallback keeps the same double-buffered
// structure with synchronous copies.
// ---------------------------------------------------------------------------
#if defined(__has_builtin)
#  if __has_builtin(__builtin_amdgcn_global_load_async_to_lds_b128)
#    define HAVE_ASYNC 1
#  endif
#endif
#ifndef HAVE_ASYNC
#  define HAVE_ASYNC 0
#endif

#if HAVE_ASYNC
typedef __attribute__((address_space(1))) v4i as1_v4i;
typedef __attribute__((address_space(3))) v4i as3_v4i;
#  if __has_builtin(__builtin_amdgcn_s_wait_asynccnt)
#    define WAIT_ASYNC(n) __builtin_amdgcn_s_wait_asynccnt(n)
#  else
#    define WAIT_ASYNC(n) asm volatile("s_wait_asynccnt %0" ::"n"(n) : "memory")
#  endif
#else
#  define WAIT_ASYNC(n) ((void)0)
#endif

// Copy 16 bytes (8 bf16) global -> LDS.
static __device__ __forceinline__ void cp16_g2l(const bf16_t* g, bf16_t* l) {
#if HAVE_ASYNC
  __builtin_amdgcn_global_load_async_to_lds_b128(
      (as1_v4i*)(void*)g, (as3_v4i*)(void*)l, 0, 0);
#else
  *(v8bf*)l = *(const v8bf*)g;
#endif
}

// ---------------------------------------------------------------------------
// WMMA fragment loaders (ISA 7.12.2 layouts, wave32)
// A-frag (16x32 bf16): lane L<16 -> row=L, K {0..7,16..23}; L>=16 -> K {8..15,24..31}
// ---------------------------------------------------------------------------
static __device__ __forceinline__ v16bf load_frag_row(const bf16_t* base, int row0,
                                                      int stride, int kbase) {
  const int lane = threadIdx.x & 31;
  const int half = lane >> 4;
  const bf16_t* p = base + (size_t)(row0 + (lane & 15)) * stride + kbase;
  union { v16bf v; v8bf h[2]; } u;
  u.h[0] = *(const v8bf*)(p + half * 8);
  u.h[1] = *(const v8bf*)(p + 16 + half * 8);
  return u.v;
}

// B-frag (32x16 bf16) from K-contiguous storage: lane L<16 -> col=L, K {0..7,16..23};
// L>=16 -> col=L-16, K {8..15,24..31}. `base` is [col][k] with row stride `stride`.
static __device__ __forceinline__ v16bf load_frag_col(const bf16_t* base, int col0,
                                                      int stride, int kbase) {
  const int lane = threadIdx.x & 31;
  const bf16_t* p = base + (size_t)(col0 + (lane & 15)) * stride + kbase + (lane >> 4) * 8;
  union { v16bf v; v8bf h[2]; } u;
  u.h[0] = *(const v8bf*)(p);
  u.h[1] = *(const v8bf*)(p + 16);
  return u.v;
}

// ---------------------------------------------------------------------------
// Weight transpose + fp32->bf16 convert:  in[K][N] fp32  ->  out[N][K] bf16
// ---------------------------------------------------------------------------
__global__ __launch_bounds__(256) void transpose_cvt_kernel(const float* __restrict__ in,
                                                            bf16_t* __restrict__ out,
                                                            int K, int N) {
  __shared__ float tile[32][33];
  const int nb = N >> 5;
  const int bn = blockIdx.x % nb;
  const int bk = blockIdx.x / nb;
  const int n0 = bn * 32, k0 = bk * 32;
  const int c = threadIdx.x & 31;
  const int r = threadIdx.x >> 5;   // 8 rows per pass
  for (int rr = 0; rr < 32; rr += 8)
    tile[r + rr][c] = in[(size_t)(k0 + r + rr) * N + n0 + c];
  __syncthreads();
  for (int rr = 0; rr < 32; rr += 8)
    out[(size_t)(n0 + r + rr) * K + k0 + c] = f2bf(tile[c][r + rr]);
}

// ---------------------------------------------------------------------------
// LayerNorm (row of 768) -> bf16
// ---------------------------------------------------------------------------
__global__ __launch_bounds__(256) void layernorm_bf16_kernel(const float* __restrict__ x,
                                                             const float* __restrict__ g,
                                                             const float* __restrict__ b,
                                                             bf16_t* __restrict__ out) {
  __shared__ float red[256];
  const int tid = threadIdx.x;
  const int row = blockIdx.x;
  const float* xr = x + (size_t)row * DD;
  float v0 = xr[tid], v1 = xr[tid + 256], v2 = xr[tid + 512];
  red[tid] = v0 + v1 + v2;
  __syncthreads();
  for (int o = 128; o > 0; o >>= 1) {
    if (tid < o) red[tid] += red[tid + o];
    __syncthreads();
  }
  const float mu = red[0] * (1.0f / 768.0f);
  __syncthreads();
  float d0 = v0 - mu, d1 = v1 - mu, d2 = v2 - mu;
  red[tid] = d0 * d0 + d1 * d1 + d2 * d2;
  __syncthreads();
  for (int o = 128; o > 0; o >>= 1) {
    if (tid < o) red[tid] += red[tid + o];
    __syncthreads();
  }
  const float rs = rsqrtf(red[0] * (1.0f / 768.0f) + 1e-5f);
  bf16_t* orow = out + (size_t)row * DD;
  orow[tid]       = f2bf(d0 * rs * g[tid]       + b[tid]);
  orow[tid + 256] = f2bf(d1 * rs * g[tid + 256] + b[tid + 256]);
  orow[tid + 512] = f2bf(d2 * rs * g[tid + 512] + b[tid + 512]);
}

// ---------------------------------------------------------------------------
// Tiled bf16 WMMA GEMM: out[M,N] = epilogue(A[M,K] @ Wt[N,K]^T + bias)
// BM=128 BN=128 BK=32, 256 threads (8 waves), wave tile 32x64 (2x4 WMMA frags)
// Double-buffered LDS with async global->LDS staging (ASYNCcnt).
// Per wave each tile is exactly 4 async b128 instructions (A:2 + B:2), so
// after issuing tile k+1, s_wait_asynccnt<=4 guarantees tile k has landed.
// EPI: 0 = bias -> fp32 ; 1 = bias + residual -> fp32 ; 2 = bias + gelu -> bf16
// ---------------------------------------------------------------------------
template <int EPI>
__global__ __launch_bounds__(256) void gemm_bf16_kernel(const bf16_t* __restrict__ A,
                                                        const bf16_t* __restrict__ Wt,
                                                        const float* __restrict__ bias,
                                                        const float* __restrict__ res,
                                                        float* __restrict__ outF,
                                                        bf16_t* __restrict__ outB,
                                                        int M, int N, int K) {
  __shared__ __align__(16) bf16_t sA[2][128 * 40];  // padded stride 40 (80B, 16B aligned)
  __shared__ __align__(16) bf16_t sB[2][128 * 40];
  const int tid = threadIdx.x;
  const int nb = N >> 7;
  const int n0 = (blockIdx.x % nb) * 128;
  const int m0 = (blockIdx.x / nb) * 128;
  const int wave = tid >> 5;
  const int wm = (wave & 3) * 32;    // wave row offset in tile
  const int wn = (wave >> 2) * 64;   // wave col offset in tile
  const int lrow = tid >> 1;         // 0..127
  const int lk = (tid & 1) * 16;     // 0 or 16

  v8f acc[2][4];
  for (int i = 0; i < 2; ++i)
    for (int j = 0; j < 4; ++j)
      for (int e = 0; e < 8; ++e) acc[i][j][e] = 0.0f;

  const bf16_t* Ag = A + (size_t)m0 * K + (size_t)lrow * K + lk;
  const bf16_t* Wg = Wt + (size_t)n0 * K + (size_t)lrow * K + lk;
  const int ldsoff = lrow * 40 + lk;
  const int nk = K >> 5;

  // stage tile kt into buffer bsel
  auto stage = [&](int kt, int bsel) {
    const bf16_t* ga = Ag + (kt << 5);
    const bf16_t* gw = Wg + (kt << 5);
    cp16_g2l(ga,     &sA[bsel][ldsoff]);
    cp16_g2l(ga + 8, &sA[bsel][ldsoff + 8]);
    cp16_g2l(gw,     &sB[bsel][ldsoff]);
    cp16_g2l(gw + 8, &sB[bsel][ldsoff + 8]);
  };

  stage(0, 0);
  int buf = 0;
  for (int kt = 0; kt < nk; ++kt) {
    if (kt + 1 < nk) {
      stage(kt + 1, buf ^ 1);   // prefetch next tile into the other buffer
      WAIT_ASYNC(4);            // current tile's 4 async loads complete
    } else {
      WAIT_ASYNC(0);
    }
    __syncthreads();            // all waves' tile data visible

    v16bf af[2], bfm[4];
    for (int i = 0; i < 2; ++i) af[i] = load_frag_row(sA[buf], wm + i * 16, 40, 0);
    for (int j = 0; j < 4; ++j) bfm[j] = load_frag_col(sB[buf], wn + j * 16, 40, 0);
    for (int i = 0; i < 2; ++i)
      for (int j = 0; j < 4; ++j)
        acc[i][j] = __builtin_amdgcn_wmma_f32_16x16x32_bf16(
            false, af[i], false, bfm[j], (short)0, acc[i][j], false, false);
    __syncthreads();            // reads done before this buffer is re-staged
    buf ^= 1;
  }

  const int lane = tid & 31;
  const int colq = lane & 15;
  const int rowh = (lane >> 4) * 8;
  for (int j = 0; j < 4; ++j) {
    const int n = n0 + wn + j * 16 + colq;
    const float bv = bias[n];
    for (int i = 0; i < 2; ++i) {
      for (int e = 0; e < 8; ++e) {
        const int m = m0 + wm + i * 16 + rowh + e;
        const size_t idx = (size_t)m * N + n;
        float v = acc[i][j][e] + bv;
        if constexpr (EPI == 0) {
          outF[idx] = v;
        } else if constexpr (EPI == 1) {
          outF[idx] = v + res[idx];
        } else {
          // tanh-approx GELU (jax.nn.gelu default)
          float gl = 0.5f * v * (1.0f + tanhf(0.7978845608f * (v + 0.044715f * v * v * v)));
          outB[idx] = f2bf(gl);
        }
      }
    }
  }
}

// ---------------------------------------------------------------------------
// Flash attention: one block = (b, h, 64-query tile), 128 threads (4 waves).
// Each wave owns 16 query rows. Online softmax; all matmuls via WMMA bf16.
// qkv layout: row m = b*N + n, col = sel*768 + h*64 + d  (fp32).
// Writes ctx as bf16 [M, 768].
// ---------------------------------------------------------------------------
__global__ __launch_bounds__(128) void attention_kernel(const float* __restrict__ qkv,
                                                        const float* __restrict__ rel_bias,
                                                        bf16_t* __restrict__ ctx) {
  __shared__ __align__(16) bf16_t sQ[64 * 72];
  __shared__ __align__(16) bf16_t sK[64 * 72];
  __shared__ __align__(16) bf16_t sVT[64 * 72];       // V transposed: [dh][key]
  __shared__ __align__(16) bf16_t sP[4 * 16 * 72];    // per-wave P tiles

  const int tid  = threadIdx.x;
  const int lane = tid & 31;
  const int wave = tid >> 5;
  const int qb = blockIdx.x & 15;
  const int h  = (blockIdx.x >> 4) % HH;
  const int b  = blockIdx.x / (16 * HH);

  const size_t rowbase = (size_t)b * NN;
  const int colq = lane & 15;
  const int rowh = (lane >> 4) * 8;

  // ---- load Q tile (64 rows x 64 dh), cvt fp32 -> bf16
  {
    const int r = tid >> 1, c0 = (tid & 1) * 32;
    const float* gp = qkv + (rowbase + qb * 64 + r) * QKVN + h * DH + c0;
    bf16_t* sp = sQ + r * 72 + c0;
    for (int i = 0; i < 32; i += 4) {
      float4 f = *(const float4*)(gp + i);
      sp[i] = f2bf(f.x); sp[i + 1] = f2bf(f.y); sp[i + 2] = f2bf(f.z); sp[i + 3] = f2bf(f.w);
    }
  }

  float m_i[8], l_i[8];
  v8f o[4];
  for (int e = 0; e < 8; ++e) { m_i[e] = -1e30f; l_i[e] = 0.0f; }
  for (int d = 0; d < 4; ++d)
    for (int e = 0; e < 8; ++e) o[d][e] = 0.0f;

  for (int kb = 0; kb < 16; ++kb) {
    __syncthreads();   // prev-iter LDS reads done (also makes Q visible on iter 0)
    // ---- load K tile row-major, V tile transposed
    {
      const int r = tid >> 1, c0 = (tid & 1) * 32;
      const float* kg = qkv + (rowbase + kb * 64 + r) * QKVN + DD + h * DH + c0;
      bf16_t* sp = sK + r * 72 + c0;
      for (int i = 0; i < 32; i += 4) {
        float4 f = *(const float4*)(kg + i);
        sp[i] = f2bf(f.x); sp[i + 1] = f2bf(f.y); sp[i + 2] = f2bf(f.z); sp[i + 3] = f2bf(f.w);
      }
      const float* vg = qkv + (rowbase + kb * 64 + r) * QKVN + 2 * DD + h * DH + c0;
      for (int i = 0; i < 32; ++i) sVT[(c0 + i) * 72 + r] = f2bf(vg[i]);
    }
    __syncthreads();

    // ---- S = Q @ K^T (per wave: 16 rows x 64 keys, K-dim 64 -> 2 WMMA each)
    const v16bf aq0 = load_frag_row(sQ, wave * 16, 72, 0);
    const v16bf aq1 = load_frag_row(sQ, wave * 16, 72, 32);
    float sv[4][8];
    for (int j = 0; j < 4; ++j) {
      v8f sa;
      for (int e = 0; e < 8; ++e) sa[e] = 0.0f;
      sa = __builtin_amdgcn_wmma_f32_16x16x32_bf16(false, aq0, false,
             load_frag_col(sK, j * 16, 72, 0), (short)0, sa, false, false);
      sa = __builtin_amdgcn_wmma_f32_16x16x32_bf16(false, aq1, false,
             load_frag_col(sK, j * 16, 72, 32), (short)0, sa, false, false);
      for (int e = 0; e < 8; ++e) {
        const int key  = kb * 64 + j * 16 + colq;
        const int qrow = qb * 64 + wave * 16 + rowh + e;
        const float bias = rel_bias[((size_t)h * NN + qrow) * NN + key];
        sv[j][e] = sa[e] * 0.125f + bias;   // scale = 1/sqrt(64)
      }
    }

    // ---- online softmax (row reductions across the 16-lane half)
    float mn[8], alpha[8], rsum[8];
    for (int e = 0; e < 8; ++e) {
      float mx = sv[0][e];
      for (int j = 1; j < 4; ++j) mx = fmaxf(mx, sv[j][e]);
      for (int msk = 1; msk < 16; msk <<= 1) mx = fmaxf(mx, __shfl_xor(mx, msk, 32));
      mn[e] = fmaxf(m_i[e], mx);
      alpha[e] = __expf(m_i[e] - mn[e]);
      m_i[e] = mn[e];
      rsum[e] = 0.0f;
    }
    bf16_t* sPw = sP + wave * 16 * 72;
    for (int j = 0; j < 4; ++j)
      for (int e = 0; e < 8; ++e) {
        const float p = __expf(sv[j][e] - mn[e]);
        rsum[e] += p;
        sPw[(rowh + e) * 72 + j * 16 + colq] = f2bf(p);
      }
    for (int e = 0; e < 8; ++e) {
      float rsv = rsum[e];
      for (int msk = 1; msk < 16; msk <<= 1) rsv += __shfl_xor(rsv, msk, 32);
      l_i[e] = l_i[e] * alpha[e] + rsv;
    }
    for (int d = 0; d < 4; ++d)
      for (int e = 0; e < 8; ++e) o[d][e] *= alpha[e];

    // ---- O += P @ V  (same-wave LDS is in-order: no barrier needed for sPw)
    const v16bf ap0 = load_frag_row(sPw, 0, 72, 0);
    const v16bf ap1 = load_frag_row(sPw, 0, 72, 32);
    for (int d = 0; d < 4; ++d) {
      o[d] = __builtin_amdgcn_wmma_f32_16x16x32_bf16(false, ap0, false,
               load_frag_col(sVT, d * 16, 72, 0), (short)0, o[d], false, false);
      o[d] = __builtin_amdgcn_wmma_f32_16x16x32_bf16(false, ap1, false,
               load_frag_col(sVT, d * 16, 72, 32), (short)0, o[d], false, false);
    }
  }

  // ---- normalize and write ctx bf16
  for (int e = 0; e < 8; ++e) {
    const float inv = 1.0f / l_i[e];
    const size_t m = rowbase + qb * 64 + wave * 16 + rowh + e;
    for (int d = 0; d < 4; ++d)
      ctx[m * DD + h * DH + d * 16 + colq] = f2bf(o[d][e] * inv);
  }
}

// ---------------------------------------------------------------------------
// Host: orchestrate the layer. All launches on `stream`, scratch from d_ws.
// ---------------------------------------------------------------------------
extern "C" void kernel_launch(void* const* d_in, const int* in_sizes, int n_in,
                              void* d_out, int out_size, void* d_ws, size_t ws_size,
                              hipStream_t stream) {
  (void)in_sizes; (void)n_in; (void)out_size; (void)ws_size;
  const float* x        = (const float*)d_in[0];
  const float* ln1_g    = (const float*)d_in[1];
  const float* ln1_b    = (const float*)d_in[2];
  const float* wqkv     = (const float*)d_in[3];
  const float* bqkv     = (const float*)d_in[4];
  const float* wo       = (const float*)d_in[5];
  const float* bo       = (const float*)d_in[6];
  const float* rel_bias = (const float*)d_in[7];
  const float* ln2_g    = (const float*)d_in[8];
  const float* ln2_b    = (const float*)d_in[9];
  const float* w1       = (const float*)d_in[10];
  const float* b1       = (const float*)d_in[11];
  const float* w2       = (const float*)d_in[12];
  const float* b2       = (const float*)d_in[13];
  float* out = (float*)d_out;

  char* ws = (char*)d_ws;
  size_t off = 0;
  auto alloc = [&](size_t bytes) -> void* {
    void* p = ws + off;
    off += (bytes + 255) & ~(size_t)255;
    return p;
  };
  bf16_t* wt_qkv = (bf16_t*)alloc((size_t)QKVN * DD * 2);   // [2304][768]
  bf16_t* wt_o   = (bf16_t*)alloc((size_t)DD * DD * 2);     // [768][768]
  bf16_t* wt_1   = (bf16_t*)alloc((size_t)DFF * DD * 2);    // [3072][768]
  bf16_t* wt_2   = (bf16_t*)alloc((size_t)DD * DFF * 2);    // [768][3072]
  bf16_t* hb     = (bf16_t*)alloc((size_t)MTOT * DD * 2);
  float*  qkvf   = (float*) alloc((size_t)MTOT * QKVN * 4);
  bf16_t* ctxb   = (bf16_t*)alloc((size_t)MTOT * DD * 2);
  float*  x2     = (float*) alloc((size_t)MTOT * DD * 4);
  bf16_t* h2b    = (bf16_t*)alloc((size_t)MTOT * DD * 2);
  bf16_t* a1b    = (bf16_t*)alloc((size_t)MTOT * DFF * 2);

  // Weights -> bf16, transposed to [N][K]
  transpose_cvt_kernel<<<(QKVN / 32) * (DD / 32), 256, 0, stream>>>(wqkv, wt_qkv, DD, QKVN);
  transpose_cvt_kernel<<<(DD / 32) * (DD / 32), 256, 0, stream>>>(wo, wt_o, DD, DD);
  transpose_cvt_kernel<<<(DFF / 32) * (DD / 32), 256, 0, stream>>>(w1, wt_1, DD, DFF);
  transpose_cvt_kernel<<<(DD / 32) * (DFF / 32), 256, 0, stream>>>(w2, wt_2, DFF, DD);

  // h = LN1(x) -> bf16
  layernorm_bf16_kernel<<<MTOT, 256, 0, stream>>>(x, ln1_g, ln1_b, hb);
  // qkv = h @ wqkv + bqkv  (fp32)
  gemm_bf16_kernel<0><<<(MTOT / 128) * (QKVN / 128), 256, 0, stream>>>(
      hb, wt_qkv, bqkv, nullptr, qkvf, nullptr, MTOT, QKVN, DD);
  // attention -> ctx bf16
  attention_kernel<<<BB * HH * (NN / 64), 128, 0, stream>>>(qkvf, rel_bias, ctxb);
  // x2 = ctx @ wo + bo + x
  gemm_bf16_kernel<1><<<(MTOT / 128) * (DD / 128), 256, 0, stream>>>(
      ctxb, wt_o, bo, x, x2, nullptr, MTOT, DD, DD);
  // h2 = LN2(x2) -> bf16
  layernorm_bf16_kernel<<<MTOT, 256, 0, stream>>>(x2, ln2_g, ln2_b, h2b);
  // a1 = gelu(h2 @ w1 + b1) -> bf16
  gemm_bf16_kernel<2><<<(MTOT / 128) * (DFF / 128), 256, 0, stream>>>(
      h2b, wt_1, b1, nullptr, nullptr, a1b, MTOT, DFF, DD);
  // out = a1 @ w2 + b2 + x2
  gemm_bf16_kernel<1><<<(MTOT / 128) * (DD / 128), 256, 0, stream>>>(
      a1b, wt_2, b2, x2, out, nullptr, MTOT, DD, DFF);
}
